// Head_2302102470842
// MI455X (gfx1250) — matrix operand order repaired and checked by reading, not compiled, not run on hardware
//
#include <hip/hip_runtime.h>

typedef __attribute__((ext_vector_type(16))) _Float16 v16h;
typedef __attribute__((ext_vector_type(8)))  float    v8f;
typedef __attribute__((ext_vector_type(8)))  _Float16 h8;
typedef __attribute__((ext_vector_type(4)))  _Float16 h4;
typedef __attribute__((ext_vector_type(2)))  _Float16 h2;

#define LDP 72   // padded leading dim (halfs) for row-major LDS tiles (bank-conflict-free b128)

// D = A(16x32 f16) * B(32x16 f16) + C(16x16 f32)
__device__ __forceinline__ v8f wmma_f16(v16h a, v16h b, v8f c) {
  return __builtin_amdgcn_wmma_f32_16x16x32_f16(false, a, false, b, (short)0, c, false, false);
}

union Frag { v16h v; h8 h[2]; };

// packed f32 pair -> f16 pair (v_cvt_pk_f16_f32)
__device__ __forceinline__ h2 pk2(float a, float b) {
  auto t = __builtin_amdgcn_cvt_pkrtz(a, b);
  h2 r; r[0] = t[0]; r[1] = t[1];
  return r;
}

// A-operand fragment (16x32) from row-major [ld=LDP] storage.
// lane -> M = lane&15; halfs [0..15] = K 16*(lane>>4)+0..15
__device__ __forceinline__ v16h load_a_frag(const _Float16* p, int row0, int col0) {
  const int lane = threadIdx.x & 31;
  const _Float16* base = p + (row0 + (lane & 15)) * LDP + col0 + (lane >> 4) * 16;
  Frag f;
  f.h[0] = *(const h8*)(base);
  f.h[1] = *(const h8*)(base + 8);
  return f.v;
}

// B-operand fragment (32x16) from row-major storage of B-TRANSPOSED [ld=LDP].
// lane -> N = lane&15, K = k0 + 16*(lane>>4) + 0..15  (one contiguous row of B^T)
__device__ __forceinline__ v16h load_bT_frag(const _Float16* pT, int n0, int k0) {
  const int lane = threadIdx.x & 31;
  const _Float16* base = pT + (n0 + (lane & 15)) * LDP + k0 + (lane >> 4) * 16;
  Frag f;
  f.h[0] = *(const h8*)(base);
  f.h[1] = *(const h8*)(base + 8);
  return f.v;
}

// Prebuilt fragment slab (global or LDS): 32 lanes x 16 halfs contiguous per lane.
__device__ __forceinline__ v16h load_frag_slab(const _Float16* p) {
  const int lane = threadIdx.x & 31;
  Frag f;
  f.h[0] = *(const h8*)(p + lane * 16);
  f.h[1] = *(const h8*)(p + lane * 16 + 8);
  return f.v;
}

// C/D 16x16 f32: VGPR r -> M = r + 8*(lane>>4), N = lane&15
__device__ __forceinline__ void store_d_f16(_Float16* p, v8f d, int row0, int col0) {
  const int lane = threadIdx.x & 31;
  const int n = lane & 15, hh = lane >> 4;
#pragma unroll
  for (int r = 0; r < 8; ++r)
    p[(row0 + r + 8 * hh) * LDP + col0 + n] = (_Float16)d[r];
}

// Transposed D store: 8 rows land at consecutive columns -> packed converts + one b128 store.
__device__ __forceinline__ void store_dT_f16(_Float16* p, v8f d, int row0, int col0) {
  const int lane = threadIdx.x & 31;
  const int n = lane & 15, hh = lane >> 4;
  union { h8 v; h2 q[4]; } t;
#pragma unroll
  for (int r = 0; r < 4; ++r) t.q[r] = pk2(d[2 * r], d[2 * r + 1]);
  *(h8*)(p + (col0 + n) * LDP + row0 + 8 * hh) = t.v;
}

// ---- setup kernel: weight f32 -> f16 WMMA B-fragments, ONCE per launch ----
// layout: wfrag[(m*8 + fid)*512 + lane*16 + e], fid = kk*4 + nt
__global__ __launch_bounds__(128) void build_wfrag_kernel(
    const float* __restrict__ Wk, const float* __restrict__ Wq,
    const float* __restrict__ Wv, _Float16* __restrict__ wfrag) {
  const int tid = threadIdx.x, wave = tid >> 5, lane = tid & 31;
  const int n = lane & 15, kh = (lane >> 4) * 16;
  const float* Ws[3] = {Wk, Wq, Wv};
#pragma unroll
  for (int m = 0; m < 3; ++m) {
#pragma unroll
    for (int j = 0; j < 2; ++j) {
      const int fid = wave + 4 * j;            // 0..7 -> kk = fid>>2, nt = fid&3
      const int kk = fid >> 2, nt = fid & 3;
      const float* w = Ws[m] + (32 * kk + kh) * 64 + 16 * nt + n;
      union { h8 v; h2 q[4]; } lo, hi;
#pragma unroll
      for (int pr = 0; pr < 8; ++pr) {
        const h2 p2 = pk2(w[(2 * pr) * 64], w[(2 * pr + 1) * 64]);
        if (pr < 4) lo.q[pr] = p2; else hi.q[pr - 4] = p2;
      }
      _Float16* dst = wfrag + ((m * 8 + fid) * 32 + lane) * 16;
      *(h8*)dst       = lo.v;
      *(h8*)(dst + 8) = hi.v;
    }
  }
}

__global__ __launch_bounds__(128) void attn_head_kernel(
    const float* __restrict__ x, const _Float16* __restrict__ wfrag,
    float* __restrict__ out) {
  __shared__ __align__(16) _Float16 sX [64 * LDP];   // x -> q -> P (per-wave stripes)
  __shared__ __align__(16) _Float16 sK [64 * LDP];   // k row-major (= B^T for q.k^T)
  __shared__ __align__(16) _Float16 sVT[64 * LDP];   // v^T row-major (= B^T for P.v)

  const int b    = blockIdx.x;
  const int tid  = threadIdx.x;
  const int wave = tid >> 5, lane = tid & 31;
  const int tbase = wave * 16;
  const int n = lane & 15, hh = lane >> 4;

  // ---- stage x: f32 -> f16 (packed cvt) into padded row-major LDS ----
  const float4* xs = (const float4*)(x + (size_t)b * 4096);
#pragma unroll
  for (int j = 0; j < 8; ++j) {
    const int i = tid + 128 * j;
    const float4 f = xs[i];
    const int t = i >> 4, c = (i & 15) * 4;
    union { h4 v; h2 q[2]; } pv;
    pv.q[0] = pk2(f.x, f.y);
    pv.q[1] = pk2(f.z, f.w);
    *(h4*)(sX + t * LDP + c) = pv.v;
  }
  __syncthreads();

  const _Float16* wfK = wfrag + 0 * 8 * 512;
  const _Float16* wfQ = wfrag + 1 * 8 * 512;
  const _Float16* wfV = wfrag + 2 * 8 * 512;

  // ---- phase 1: q,k,v projections (24 WMMAs); weight B-frags stream from L2 ----
  {
    const v16h a0 = load_a_frag(sX, tbase, 0);   // x stripe dead after this -> q reuses it
    const v16h a1 = load_a_frag(sX, tbase, 32);
#pragma unroll
    for (int nt = 0; nt < 4; ++nt) {
      v8f cq = {}, ck = {}, cv = {};
      cq = wmma_f16(a0, load_frag_slab(wfQ + nt * 512),       cq);
      cq = wmma_f16(a1, load_frag_slab(wfQ + (4 + nt) * 512), cq);
      ck = wmma_f16(a0, load_frag_slab(wfK + nt * 512),       ck);
      ck = wmma_f16(a1, load_frag_slab(wfK + (4 + nt) * 512), ck);
      cv = wmma_f16(a0, load_frag_slab(wfV + nt * 512),       cv);
      cv = wmma_f16(a1, load_frag_slab(wfV + (4 + nt) * 512), cv);
      store_d_f16 (sX,  cq, tbase, 16 * nt);     // q (row-major, A-source for S)
      store_d_f16 (sK,  ck, tbase, 16 * nt);     // k (row-major = B^T of k^T)
      store_dT_f16(sVT, cv, tbase, 16 * nt);     // v^T, single b128 store
    }
  }
  __syncthreads();  // all waves' k / v^T visible

  // ---- phase 2: S = q @ k^T (8 WMMAs) + scale + causal mask + softmax in registers ----
  v8f sacc[4];
  {
    const v16h a0 = load_a_frag(sX, tbase, 0);
    const v16h a1 = load_a_frag(sX, tbase, 32);
#pragma unroll
    for (int st = 0; st < 4; ++st) {
      v8f c = {};
      c = wmma_f16(a0, load_bT_frag(sK, 16 * st, 0),  c);
      c = wmma_f16(a1, load_bT_frag(sK, 16 * st, 32), c);
      sacc[st] = c;
    }
  }

  _Float16* sP = sX + tbase * LDP;               // reuse stripe for P
  const float NEG_INF = -__builtin_inff();
#pragma unroll
  for (int r = 0; r < 8; ++r) {
    const int t = tbase + r + 8 * hh;
    float vals[4];
    float mx = -3.0e38f;
#pragma unroll
    for (int st = 0; st < 4; ++st) {
      const int s = 16 * st + n;
      float v = sacc[st][r] * 0.125f;            // * C^-0.5
      if (s > t) v = NEG_INF;                    // causal mask
      vals[st] = v;
      mx = fmaxf(mx, v);
    }
#pragma unroll
    for (int off = 1; off < 16; off <<= 1)       // row lives in a 16-lane half
      mx = fmaxf(mx, __shfl_xor(mx, off, 32));
    float sum = 0.f;
#pragma unroll
    for (int st = 0; st < 4; ++st) {
      const float pv = __expf(vals[st] - mx);
      vals[st] = pv;
      sum += pv;
    }
#pragma unroll
    for (int off = 1; off < 16; off <<= 1)
      sum += __shfl_xor(sum, off, 32);
    const float inv = 1.0f / sum;
#pragma unroll
    for (int st = 0; st < 4; ++st)
      sP[(r + 8 * hh) * LDP + 16 * st + n] = (_Float16)(vals[st] * inv);
  }
  // P is written/read by this wave only; per-wave DS ordering suffices, no barrier.

  // ---- phase 3: O = P @ V (8 WMMAs) -> global f32, coalesced per half-wave ----
  {
    const v16h a0 = load_a_frag(sX, tbase, 0);
    const v16h a1 = load_a_frag(sX, tbase, 32);
    float* ob = out + (size_t)b * 4096;
#pragma unroll
    for (int nt = 0; nt < 4; ++nt) {
      v8f c = {};
      c = wmma_f16(a0, load_bT_frag(sVT, 16 * nt, 0),  c);
      c = wmma_f16(a1, load_bT_frag(sVT, 16 * nt, 32), c);
#pragma unroll
      for (int r = 0; r < 8; ++r)
        ob[(tbase + r + 8 * hh) * 64 + 16 * nt + n] = c[r];
    }
  }
}

extern "C" void kernel_launch(void* const* d_in, const int* in_sizes, int n_in,
                              void* d_out, int out_size, void* d_ws, size_t ws_size,
                              hipStream_t stream) {
  const float* x  = (const float*)d_in[0];
  const float* Wk = (const float*)d_in[1];
  const float* Wq = (const float*)d_in[2];
  const float* Wv = (const float*)d_in[3];
  float* out = (float*)d_out;
  _Float16* wfrag = (_Float16*)d_ws;            // needs 3*8*512*2 = 24 KB of scratch

  build_wfrag_kernel<<<1, 128, 0, stream>>>(Wk, Wq, Wv, wfrag);

  const int nbatch = in_sizes[0] / (64 * 64);   // B = 16384
  attn_head_kernel<<<nbatch, 128, 0, stream>>>(x, wfrag, out);
}
